// BioGuardPretrain_91250875171516
// MI455X (gfx1250) — matrix operand order repaired and checked by reading, not compiled, not run on hardware
//
#include <hip/hip_runtime.h>
#include <hip/hip_bf16.h>

typedef __attribute__((ext_vector_type(16))) _Float16 v16h;
typedef __attribute__((ext_vector_type(8)))  _Float16 v8h;
typedef __attribute__((ext_vector_type(8)))  float    v8f;

#define NODE_DIM 32
#define EDGE_DIM 16
#define DDIM 128
#define HHEADS 4
#define CCH 128
#define HC 512
#define OUTD 24

__device__ __forceinline__ v16h combine8(v8h lo, v8h hi) {
    v16h a;
#pragma unroll
    for (int i = 0; i < 8; ++i) { a[i] = lo[i]; a[8 + i] = hi[i]; }
    return a;
}

__device__ __forceinline__ void atomicMaxF(float* addr, float val) {
    unsigned int* ua = (unsigned int*)addr;
    unsigned int old = __atomic_load_n(ua, __ATOMIC_RELAXED);
    while (true) {
        float of = __uint_as_float(old);
        if (of >= val) break;
        unsigned int assumed = old;
        old = atomicCAS(ua, assumed, __float_as_uint(val));
        if (old == assumed) break;
    }
}

// ---------------------------------------------------------------------------
// Kernel 0: weight prep — transpose to f16 K-major (fragment-friendly) layout,
// fold BatchNorm into per-channel scale/shift.
// Region element counts:
//   Wenc_t: 128*32   Wl_t: 512*128   Wr_t: 512*128   We_t: 512*32 (K padded)
//   Wp_t:   32*512 (N padded)        sbuf/c2buf: 128
// ---------------------------------------------------------------------------
__global__ void prep_kernel(const float* __restrict__ W_enc, const float* __restrict__ Wl,
                            const float* __restrict__ Wr,    const float* __restrict__ We,
                            const float* __restrict__ Wp,    const float* __restrict__ b_enc,
                            const float* __restrict__ bn_g,  const float* __restrict__ bn_b,
                            const float* __restrict__ bn_m,  const float* __restrict__ bn_v,
                            _Float16* __restrict__ Wenc_t, _Float16* __restrict__ Wl_t,
                            _Float16* __restrict__ Wr_t,   _Float16* __restrict__ We_t,
                            _Float16* __restrict__ Wp_t,
                            float* __restrict__ sbuf, float* __restrict__ c2buf)
{
    int i = blockIdx.x * blockDim.x + threadIdx.x;
    const int R0 = 128 * 32;
    const int R1 = R0 + 512 * 128;
    const int R2 = R1 + 512 * 128;
    const int R3 = R2 + 512 * 32;
    const int R4 = R3 + 32 * 512;
    const int R5 = R4 + 128;
    if (i < R0) {                       // Wenc_t[c*32+k] = W_enc[k*128+c]
        int c = i >> 5, k = i & 31;
        Wenc_t[i] = (_Float16)W_enc[k * 128 + c];
    } else if (i < R1) {                // Wl_t[n*128+k] = Wl[k*512+n]
        int j = i - R0; int n = j >> 7, k = j & 127;
        Wl_t[j] = (_Float16)Wl[k * 512 + n];
    } else if (i < R2) {
        int j = i - R1; int n = j >> 7, k = j & 127;
        Wr_t[j] = (_Float16)Wr[k * 512 + n];
    } else if (i < R3) {                // We_t[n*32+k], zero-pad k>=16
        int j = i - R2; int n = j >> 5, k = j & 31;
        We_t[j] = (k < EDGE_DIM) ? (_Float16)We[k * 512 + n] : (_Float16)0.f;
    } else if (i < R4) {                // Wp_t[n*512+k], zero-pad n>=24
        int j = i - R3; int n = j >> 9, k = j & 511;
        Wp_t[j] = (n < OUTD) ? (_Float16)Wp[k * OUTD + n] : (_Float16)0.f;
    } else if (i < R5) {                // fold BN: y = (x@W)*s + c2, then ReLU
        int j = i - R4;
        float s = bn_g[j] * rsqrtf(bn_v[j] + 1e-5f);
        sbuf[j] = s;
        c2buf[j] = (b_enc[j] - bn_m[j]) * s + bn_b[j];
    }
}

__global__ void init_kernel(float* __restrict__ out_hc, float* __restrict__ segmax,
                            float* __restrict__ segsum, long long nhc, long long n4)
{
    long long i = (long long)blockIdx.x * blockDim.x + threadIdx.x;
    if (i < nhc) out_hc[i] = 0.f;
    if (i < n4) { segmax[i] = -3.0e38f; segsum[i] = 0.f; }
}

// ---------------------------------------------------------------------------
// Kernel 1: node pipeline. Per wave: 16 nodes.
//   h = relu((x @ W_enc)*s + c2)   via WMMA (K=32)
//   xl = h @ Wl + bl ; xr = h @ Wr + br   via WMMA (K=128), stored f16.
// ---------------------------------------------------------------------------
__global__ __launch_bounds__(256)
void node_gemm_kernel(const float* __restrict__ x,
                      const _Float16* __restrict__ Wenc_t,
                      const _Float16* __restrict__ Wl_t,
                      const _Float16* __restrict__ Wr_t,
                      const float* __restrict__ sbuf, const float* __restrict__ c2buf,
                      const float* __restrict__ bl,   const float* __restrict__ br,
                      _Float16* __restrict__ xl_h, _Float16* __restrict__ xr_h, int N)
{
    __shared__ _Float16 h_sh[8][16][128];   // per-wave f16 h tile, 32 KB
    const int lane  = threadIdx.x & 31;
    const int wave  = threadIdx.x >> 5;
    const int tile  = blockIdx.x * 8 + wave;
    const int m0    = tile * 16;
    if (m0 >= N) return;
    const int col16 = lane & 15;
    const int half  = lane >> 4;
    const int koff  = half * 8;    // unused; A fragment also uses koffb
    const int koffb = half * 16;   // B-fragment K base

    // encoder A fragment: 16x32 f16 from x [N,32]
    int arow = m0 + col16; if (arow >= N) arow = N - 1;
    const float* xp = x + (size_t)arow * NODE_DIM;
    v16h aenc;
#pragma unroll
    for (int i = 0; i < 8; ++i) {
        aenc[i]     = (_Float16)xp[koffb + i];
        aenc[8 + i] = (_Float16)xp[koffb + 8 + i];
    }

    // h tile (16x128) into LDS
#pragma unroll
    for (int nt = 0; nt < 8; ++nt) {
        const v16h b = *(const v16h*)(Wenc_t + (size_t)(nt * 16 + col16) * 32 + koffb);
        v8f c = {};
        c = __builtin_amdgcn_wmma_f32_16x16x32_f16(false, aenc, false, b,
                                                   (short)0, c, false, false);
        const float sv = sbuf[nt * 16 + col16];
        const float cv = c2buf[nt * 16 + col16];
#pragma unroll
        for (int r = 0; r < 8; ++r) {
            float v = c[r] * sv + cv;
            v = v > 0.f ? v : 0.f;
            h_sh[wave][r + half * 8][nt * 16 + col16] = (_Float16)v;
        }
    }
    // same-wave LDS ops are in-order: no barrier needed (private region per wave)

    // A fragments for K=128 (4 k-steps)
    v16h ah[4];
#pragma unroll
    for (int s = 0; s < 4; ++s) {
        const v8h lo = *(const v8h*)&h_sh[wave][col16][s * 32 + koffb];
        const v8h hi = *(const v8h*)&h_sh[wave][col16][s * 32 + koffb + 8];
        ah[s] = combine8(lo, hi);
    }

#pragma unroll 1
    for (int p = 0; p < 2; ++p) {
        const _Float16* W = p ? Wr_t : Wl_t;
        const float*   bv = p ? br   : bl;
        _Float16*      o  = p ? xr_h : xl_h;
#pragma unroll 1
        for (int t = 0; t < 32; ++t) {
            v8f c = {};
#pragma unroll
            for (int s = 0; s < 4; ++s) {
                const v16h b = *(const v16h*)(W + (size_t)(t * 16 + col16) * 128
                                                + s * 32 + koffb);
                c = __builtin_amdgcn_wmma_f32_16x16x32_f16(false, ah[s], false, b,
                                                           (short)0, c, false, false);
            }
            const int col = t * 16 + col16;
            const float bias = bv[col];
#pragma unroll
            for (int r = 0; r < 8; ++r) {
                const int row = m0 + r + half * 8;
                if (row < N) o[(size_t)row * HC + col] = (_Float16)(c[r] + bias);
            }
        }
    }
}

// ---------------------------------------------------------------------------
// Kernel 2: fused edge projection + attention logits + segment max.
// Per wave: 16 edges. e = edge_attr@We via WMMA (K padded to 32),
// v = leaky(xl[src]+xr[dst]+e); logits[e,h] = sum_c v*att[h,c].
// ---------------------------------------------------------------------------
__global__ __launch_bounds__(256)
void edge_logits_kernel(const float* __restrict__ ea,
                        const _Float16* __restrict__ We_t,
                        const _Float16* __restrict__ xl_h,
                        const _Float16* __restrict__ xr_h,
                        const float* __restrict__ att,
                        const int* __restrict__ srcI, const int* __restrict__ dstI,
                        float* __restrict__ logits, float* __restrict__ segmax, int E)
{
    const int lane  = threadIdx.x & 31;
    const int wave  = threadIdx.x >> 5;
    const int tile  = blockIdx.x * 8 + wave;
    const int m0    = tile * 16;
    if (m0 >= E) return;
    const int col16 = lane & 15;
    const int half  = lane >> 4;
    const int koff  = half * 8;
    const int koffb = half * 16;

    int sidx[8], didx[8];
#pragma unroll
    for (int r = 0; r < 8; ++r) {
        int er = m0 + r + half * 8; if (er >= E) er = E - 1;
        sidx[r] = srcI[er];
        didx[r] = dstI[er];
    }

    int arow = m0 + col16; if (arow >= E) arow = E - 1;
    const float* ep = ea + (size_t)arow * EDGE_DIM;
    v16h a;
#pragma unroll
    for (int i = 0; i < 8; ++i) {          // K 0..15 real (half 0), 16..31 zero pad (half 1)
        a[i]     = half ? (_Float16)0.f : (_Float16)ep[i];
        a[8 + i] = half ? (_Float16)0.f : (_Float16)ep[8 + i];
    }

    float acc[8];
#pragma unroll
    for (int r = 0; r < 8; ++r) acc[r] = 0.f;

#pragma unroll 1
    for (int t = 0; t < 32; ++t) {
        const v16h b = *(const v16h*)(We_t + (size_t)(t * 16 + col16) * 32 + koffb);
        v8f c = {};
        c = __builtin_amdgcn_wmma_f32_16x16x32_f16(false, a, false, b,
                                                   (short)0, c, false, false);
        const int col  = t * 16 + col16;
        const int head = t >> 3;
        const float attv = att[head * CCH + (col & 127)];
#pragma unroll
        for (int r = 0; r < 8; ++r) {
            float v = c[r] + (float)xl_h[(size_t)sidx[r] * HC + col]
                           + (float)xr_h[(size_t)didx[r] * HC + col];
            v = v > 0.f ? v : 0.2f * v;
            acc[r] += v * attv;
        }
        if ((t & 7) == 7) {                 // finished one head's 128 columns
#pragma unroll
            for (int r = 0; r < 8; ++r) {
#pragma unroll
                for (int off = 1; off < 16; off <<= 1)
                    acc[r] += __shfl_xor(acc[r], off, 32);
            }
            if (col16 == 0) {
#pragma unroll
                for (int r = 0; r < 8; ++r) {
                    const int er = m0 + r + half * 8;
                    if (er < E) {
                        logits[(size_t)er * HHEADS + head] = acc[r];
                        atomicMaxF(&segmax[(size_t)didx[r] * HHEADS + head], acc[r]);
                    }
                }
            }
#pragma unroll
            for (int r = 0; r < 8; ++r) acc[r] = 0.f;
        }
    }
}

__global__ void exp_sum_kernel(const float* __restrict__ logits,
                               const float* __restrict__ segmax,
                               const int* __restrict__ dstI,
                               float* __restrict__ exb, float* __restrict__ segsum, int E)
{
    int i = blockIdx.x * blockDim.x + threadIdx.x;
    if (i >= E * HHEADS) return;
    int e = i >> 2, h = i & 3;
    int d = dstI[e];
    float v = __expf(logits[i] - segmax[d * HHEADS + h]);
    exb[i] = v;
    atomicAdd(&segsum[d * HHEADS + h], v);
}

// Kernel 4: msg = xl[src] * alpha, scatter-add into out_hc (f32 atomics).
__global__ void msg_kernel(const _Float16* __restrict__ xl_h,
                           const float* __restrict__ exb, const float* __restrict__ segsum,
                           const int* __restrict__ srcI, const int* __restrict__ dstI,
                           float* __restrict__ out_hc, long long total)
{
    long long i = (long long)blockIdx.x * blockDim.x + threadIdx.x;
    if (i >= total) return;
    int e = (int)(i >> 7);
    int q = (int)(i & 127);
    int col0 = q * 4;
    int head = col0 >> 7;
    int s = srcI[e], d = dstI[e];
    float alpha = exb[(size_t)e * HHEADS + head] / segsum[(size_t)d * HHEADS + head];
    const _Float16* xp = xl_h + (size_t)s * HC + col0;
    float* op = out_hc + (size_t)d * HC + col0;
#pragma unroll
    for (int j = 0; j < 4; ++j) atomicAdd(op + j, (float)xp[j] * alpha);
}

// Kernel 5: head GEMM  d_out = (out_hc + bias_conv) @ Wp + bp   [N,24]
__global__ __launch_bounds__(256)
void head_kernel(const float* __restrict__ out_hc, const float* __restrict__ bias_conv,
                 const _Float16* __restrict__ Wp_t, const float* __restrict__ bp,
                 float* __restrict__ out, int N)
{
    const int lane  = threadIdx.x & 31;
    const int wave  = threadIdx.x >> 5;
    const int tile  = blockIdx.x * 8 + wave;
    const int m0    = tile * 16;
    if (m0 >= N) return;
    const int col16 = lane & 15;
    const int half  = lane >> 4;
    const int koff  = half * 8;
    const int koffb = half * 16;

    int arow = m0 + col16; if (arow >= N) arow = N - 1;
    const float* hp = out_hc + (size_t)arow * HC;

    v8f c0 = {}, c1 = {};
#pragma unroll 1
    for (int s = 0; s < 16; ++s) {
        v16h a;
#pragma unroll
        for (int i = 0; i < 8; ++i) {
            int k0 = s * 32 + koffb + i;
            int k1 = k0 + 8;
            a[i]     = (_Float16)(hp[k0] + bias_conv[k0]);
            a[8 + i] = (_Float16)(hp[k1] + bias_conv[k1]);
        }
        const v16h b0 = *(const v16h*)(Wp_t + (size_t)(col16)      * HC + s * 32 + koffb);
        const v16h b1 = *(const v16h*)(Wp_t + (size_t)(16 + col16) * HC + s * 32 + koffb);
        c0 = __builtin_amdgcn_wmma_f32_16x16x32_f16(false, a, false, b0, (short)0, c0, false, false);
        c1 = __builtin_amdgcn_wmma_f32_16x16x32_f16(false, a, false, b1, (short)0, c1, false, false);
    }
#pragma unroll
    for (int r = 0; r < 8; ++r) {
        const int row = m0 + r + half * 8;
        if (row < N) {
            out[(size_t)row * OUTD + col16] = c0[r] + bp[col16];
            if (col16 < 8)
                out[(size_t)row * OUTD + 16 + col16] = c1[r] + bp[16 + col16];
        }
    }
}

extern "C" void kernel_launch(void* const* d_in, const int* in_sizes, int n_in,
                              void* d_out, int out_size, void* d_ws, size_t ws_size,
                              hipStream_t stream)
{
    const float* x         = (const float*)d_in[0];
    const float* edge_attr = (const float*)d_in[1];
    const int*   edge_idx  = (const int*)  d_in[2];
    const float* W_enc     = (const float*)d_in[3];
    const float* b_enc     = (const float*)d_in[4];
    const float* bn_g      = (const float*)d_in[5];
    const float* bn_b      = (const float*)d_in[6];
    const float* bn_m      = (const float*)d_in[7];
    const float* bn_v      = (const float*)d_in[8];
    const float* Wl        = (const float*)d_in[9];
    const float* bl        = (const float*)d_in[10];
    const float* Wr        = (const float*)d_in[11];
    const float* br        = (const float*)d_in[12];
    const float* We        = (const float*)d_in[13];
    const float* att       = (const float*)d_in[14];
    const float* bias_conv = (const float*)d_in[15];
    const float* Wp        = (const float*)d_in[16];
    const float* bp        = (const float*)d_in[17];
    float* out = (float*)d_out;

    const int N = in_sizes[0] / NODE_DIM;
    const int E = in_sizes[1] / EDGE_DIM;
    const int* srcI = edge_idx;
    const int* dstI = edge_idx + E;

    // workspace partition (256B aligned)
    char* ws = (char*)d_ws;
    size_t off = 0;
    auto alloc = [&](size_t bytes) -> void* {
        void* p = ws + off;
        off += (bytes + 255) & ~(size_t)255;
        return p;
    };
    _Float16* Wenc_t = (_Float16*)alloc(128 * 32  * sizeof(_Float16));
    _Float16* Wl_t   = (_Float16*)alloc(512 * 128 * sizeof(_Float16));
    _Float16* Wr_t   = (_Float16*)alloc(512 * 128 * sizeof(_Float16));
    _Float16* We_t   = (_Float16*)alloc(512 * 32  * sizeof(_Float16));
    _Float16* Wp_t   = (_Float16*)alloc(32 * 512  * sizeof(_Float16));
    float*    sbuf   = (float*)alloc(128 * sizeof(float));
    float*    c2buf  = (float*)alloc(128 * sizeof(float));
    _Float16* xl_h   = (_Float16*)alloc((size_t)N * HC * sizeof(_Float16));
    _Float16* xr_h   = (_Float16*)alloc((size_t)N * HC * sizeof(_Float16));
    float*    logits = (float*)alloc((size_t)E * HHEADS * sizeof(float));
    float*    exb    = (float*)alloc((size_t)E * HHEADS * sizeof(float));
    float*    segmax = (float*)alloc((size_t)N * HHEADS * sizeof(float));
    float*    segsum = (float*)alloc((size_t)N * HHEADS * sizeof(float));
    float*    out_hc = (float*)alloc((size_t)N * HC * sizeof(float));

    // 0) weight prep + BN fold
    {
        int total = 128*32 + 512*128 + 512*128 + 512*32 + 32*512 + 128;
        prep_kernel<<<(total + 255) / 256, 256, 0, stream>>>(
            W_enc, Wl, Wr, We, Wp, b_enc, bn_g, bn_b, bn_m, bn_v,
            Wenc_t, Wl_t, Wr_t, We_t, Wp_t, sbuf, c2buf);
    }
    // 1) init accumulators
    {
        long long nhc = (long long)N * HC;
        long long n4  = (long long)N * HHEADS;
        init_kernel<<<(unsigned)((nhc + 255) / 256), 256, 0, stream>>>(
            out_hc, segmax, segsum, nhc, n4);
    }
    // 2) node encoder + xl/xr projections (WMMA)
    {
        int tiles = (N + 15) / 16;
        node_gemm_kernel<<<(tiles + 7) / 8, 256, 0, stream>>>(
            x, Wenc_t, Wl_t, Wr_t, sbuf, c2buf, bl, br, xl_h, xr_h, N);
    }
    // 3) edge projection + logits + segment max (WMMA + gathers)
    {
        int tiles = (E + 15) / 16;
        edge_logits_kernel<<<(tiles + 7) / 8, 256, 0, stream>>>(
            edge_attr, We_t, xl_h, xr_h, att, srcI, dstI, logits, segmax, E);
    }
    // 4) exp + segment sum
    exp_sum_kernel<<<(E * HHEADS + 255) / 256, 256, 0, stream>>>(
        logits, segmax, dstI, exb, segsum, E);
    // 5) weighted message scatter-add
    {
        long long total = (long long)E * 128;
        msg_kernel<<<(unsigned)((total + 255) / 256), 256, 0, stream>>>(
            xl_h, exb, segsum, srcI, dstI, out_hc, total);
    }
    // 6) prediction head (WMMA)
    {
        int tiles = (N + 15) / 16;
        head_kernel<<<(tiles + 7) / 8, 256, 0, stream>>>(
            out_hc, bias_conv, Wp_t, bp, out, N);
    }
}
